// ContinuousGenHyperConnections_28424093564974
// MI455X (gfx1250) — matrix operand order repaired and checked by reading, not compiled, not run on hardware
//
#include <hip/hip_runtime.h>

typedef __bf16 bf16_t;
typedef __attribute__((ext_vector_type(16))) __bf16 v16bf;
typedef __attribute__((ext_vector_type(8)))  float  v8f;

#define IN_DIM    4096
#define BLOCK     1024
#define NCOLS     41      // 4 (W_ri) + 4 (W_wo) + 1 (W_dt) + 16 (W_conv) + 16 (W_diss)
#define NCOLS_PAD 48      // 3 WMMA N-tiles of 16
#define NTILES    3
#define KTILES    128     // 4096 / 32
#define TILE_ROWS 16
#define EPSF      1.1920928955078125e-07f

// ---------------------------------------------------------------------------
// Kernel 0: convert all projection weights to bf16 and pre-swizzle them into
// the WMMA B-fragment layout (ISA 7.12.2, 16-bit B 32x16):
//   lane l: N = l&15, half = l>>4 ; element i (0..15) -> K = ktile*32 + 8*half + i + (i>=8 ? 8 : 0)
// Buffer layout: wswz[((ktile*3 + ntile)*32 + lane)*16 + i]   (384 KB in d_ws)
// ---------------------------------------------------------------------------
__global__ void __launch_bounds__(256)
cghc_weight_swizzle(const float* __restrict__ Wri, const float* __restrict__ Wwo,
                    const float* __restrict__ Wdt, const float* __restrict__ Wconv,
                    const float* __restrict__ Wdiss, bf16_t* __restrict__ wswz) {
  int idx = blockIdx.x * blockDim.x + threadIdx.x;
  const int total = KTILES * NTILES * 32 * 16;
  if (idx >= total) return;
  int i    = idx & 15;
  int l    = (idx >> 4) & 31;
  int c    = (idx >> 9) % NTILES;
  int t    = idx / (16 * 32 * NTILES);
  int half = l >> 4;
  int col  = c * 16 + (l & 15);
  int K    = t * 32 + 8 * half + i + ((i >= 8) ? 8 : 0);
  float v = 0.0f;
  if      (col < 4)   v = Wri [(col     ) * IN_DIM + K];
  else if (col < 8)   v = Wwo [(col -  4) * IN_DIM + K];
  else if (col == 8)  v = Wdt [K];
  else if (col < 25)  v = Wconv[(col - 9) * IN_DIM + K];
  else if (col < 41)  v = Wdiss[(col - 25) * IN_DIM + K];
  wswz[idx] = (bf16_t)v;
}

// ---------------------------------------------------------------------------
// Main fused kernel: one workgroup (256 thr = 8 waves) per 16 batch rows.
// ---------------------------------------------------------------------------
__global__ void __launch_bounds__(256)
cghc_kernel(const float* __restrict__ x,
            const float* __restrict__ read_in,  const float* __restrict__ alpha_ri,
            const float* __restrict__ write_out,const float* __restrict__ alpha_wo,
            const float* __restrict__ log_dt,   const float* __restrict__ b_dt,
            const float* __restrict__ conserv_A,const float* __restrict__ b_conv,
            const float* __restrict__ diss_A,   const float* __restrict__ b_diss,
            const bf16_t* __restrict__ wswz,
            float* __restrict__ out, int Brows) {
  __shared__ float hbuf[TILE_ROWS * NCOLS_PAD];   // projection results h[row][col]
  __shared__ float sums[TILE_ROWS * 16];          // sumsq partials
  __shared__ float invr[TILE_ROWS];               // 1/rms per row
  __shared__ float Gm[TILE_ROWS * 16];            // fused mix matrix per row

  const int tid      = threadIdx.x;
  const int tileBase = blockIdx.x * TILE_ROWS;
  const int row      = tid >> 4;                  // 0..15 (row within tile)
  const int sub      = tid & 15;                  // 0..15 (lane within row group)
  const int grow     = tileBase + row;            // global row

  for (int k = tid; k < TILE_ROWS * NCOLS_PAD; k += 256) hbuf[k] = 0.0f;

  // ---- Phase 1: sum of squares (coalesced float4 reads, 256B per 16-thr group)
  float ss = 0.0f;
  if (grow < Brows) {
    const float4* xg = (const float4*)(x + (size_t)grow * IN_DIM);
    for (int k = 0; k < 64; ++k) {
      float4 v = xg[sub + 16 * k];
      ss += v.x * v.x + v.y * v.y + v.z * v.z + v.w * v.w;
    }
  }
  sums[row * 16 + sub] = ss;
  __syncthreads();

  if (tid < TILE_ROWS) {
    float s = 0.0f;
    for (int k = 0; k < 16; ++k) s += sums[tid * 16 + k];
    invr[tid] = rsqrtf(s * (1.0f / IN_DIM) + EPSF);
  }
  __syncthreads();

  // ---- Phase 2: projections h = x_norm * W^T via v_wmma_f32_16x16x32_bf16.
  // Wave w owns K in [w*512, (w+1)*512). Each K-step builds ONE A fragment and
  // issues 3 WMMAs (one per 16-col N-tile) into 3 live accumulators, so the
  // f32->bf16 conversion + scaling VALU work is not replicated per N-tile.
  {
    const int wave = tid >> 5;          // 0..7
    const int lane = tid & 31;
    const int am   = lane & 15;         // A-row (M) == B-col (N)
    const int half = lane >> 4;
    const int arowi = (tileBase + am < Brows) ? (tileBase + am) : (Brows - 1);
    const float ascale = invr[am];
    const float* arow  = x + (size_t)arowi * IN_DIM;

    v8f acc0 = {}, acc1 = {}, acc2 = {};
    for (int kk = 0; kk < 16; ++kk) {
      const int kt    = wave * 16 + kk;
      const int kbase = kt * 32;
      // A fragment (ISA 16-bit A 16x32): elems 0-7 -> K=kbase+8*half+0..7,
      // elems 8-15 -> K=kbase+16+8*half+0..7  (two aligned float4 pairs)
      const float4* p0 = (const float4*)(arow + kbase + 8 * half);
      const float4* p1 = (const float4*)(arow + kbase + 16 + 8 * half);
      float4 q0 = p0[0], q1 = p0[1], q2 = p1[0], q3 = p1[1];
      v16bf a;
      a[0]  = (bf16_t)(q0.x * ascale); a[1]  = (bf16_t)(q0.y * ascale);
      a[2]  = (bf16_t)(q0.z * ascale); a[3]  = (bf16_t)(q0.w * ascale);
      a[4]  = (bf16_t)(q1.x * ascale); a[5]  = (bf16_t)(q1.y * ascale);
      a[6]  = (bf16_t)(q1.z * ascale); a[7]  = (bf16_t)(q1.w * ascale);
      a[8]  = (bf16_t)(q2.x * ascale); a[9]  = (bf16_t)(q2.y * ascale);
      a[10] = (bf16_t)(q2.z * ascale); a[11] = (bf16_t)(q2.w * ascale);
      a[12] = (bf16_t)(q3.x * ascale); a[13] = (bf16_t)(q3.y * ascale);
      a[14] = (bf16_t)(q3.z * ascale); a[15] = (bf16_t)(q3.w * ascale);
      // B fragments: one aligned 32-byte load each from the pre-swizzled buffer
      const v16bf* bb = (const v16bf*)(wswz + (size_t)(kt * NTILES) * 32 * 16);
      v16bf b0 = bb[0 * 32 + lane];
      v16bf b1 = bb[1 * 32 + lane];
      v16bf b2 = bb[2 * 32 + lane];
      acc0 = __builtin_amdgcn_wmma_f32_16x16x32_bf16(false, a, false, b0,
                                                     (short)0, acc0, false, false);
      acc1 = __builtin_amdgcn_wmma_f32_16x16x32_bf16(false, a, false, b1,
                                                     (short)0, acc1, false, false);
      acc2 = __builtin_amdgcn_wmma_f32_16x16x32_bf16(false, a, false, b2,
                                                     (short)0, acc2, false, false);
    }
    // C/D layout: VGPR r -> M = r + 8*half, N = lane&15. Reduce across waves.
    if (am < NCOLS) {                     // col 0..15 always < 41
      for (int r = 0; r < 8; ++r)
        atomicAdd(&hbuf[(r + 8 * half) * NCOLS_PAD + (0 * 16 + am)], acc0[r]);
    }
    {
      for (int r = 0; r < 8; ++r)         // col 16..31 always < 41
        atomicAdd(&hbuf[(r + 8 * half) * NCOLS_PAD + (1 * 16 + am)], acc1[r]);
    }
    if (2 * 16 + am < NCOLS) {            // col 32..40 only
      for (int r = 0; r < 8; ++r)
        atomicAdd(&hbuf[(r + 8 * half) * NCOLS_PAD + (2 * 16 + am)], acc2[r]);
    }
  }
  __syncthreads();

  // ---- Phase 3: per-row 4x4 algebra + expm (one thread per row)
  if (tid < TILE_ROWS && tileBase + tid < Brows) {
    const float* h = hbuf + tid * NCOLS_PAD;
    const float a_ri = alpha_ri[0], a_wo = alpha_wo[0];
    float rin[4], wout[4];
    for (int n = 0; n < 4; ++n) {
      rin[n]  = 1.0f / (1.0f + __expf(-(a_ri * h[n]     + read_in[n])));
      wout[n] = 2.0f / (1.0f + __expf(-(a_wo * h[4 + n] + write_out[n])));
    }
    float Mm[4][4], R[4][4], Amat[4][4];
    for (int i = 0; i < 4; ++i)
      for (int j = 0; j < 4; ++j) {
        Mm[i][j] = conserv_A[i * 4 + j] + h[9  + i * 4 + j] + b_conv[i * 4 + j];
        R [i][j] = diss_A  [i * 4 + j] + h[25 + i * 4 + j] + b_diss[i * 4 + j];
      }
    for (int i = 0; i < 4; ++i)
      for (int j = 0; j < 4; ++j) {
        float d = 0.0f;
        for (int k = 0; k < 4; ++k) d += R[i][k] * R[j][k];
        Amat[i][j] = (Mm[i][j] - Mm[j][i]) - d;
      }
    // dt = exp(log_dt) + softplus(h_dt + b_dt), clamped
    float z  = h[8] + b_dt[0];
    float sp = (z > 20.0f) ? z : log1pf(expf(z));
    float dt = expf(log_dt[0]) + sp;
    dt = fminf(fmaxf(dt, 1e-3f), 1.0f);
    // expm(dt*A): scaling-and-squaring + 12-term Taylor
    float T[4][4];
    float nrm = 0.0f;
    for (int i = 0; i < 4; ++i) {
      float rs = 0.0f;
      for (int j = 0; j < 4; ++j) { float v = dt * Amat[i][j]; T[i][j] = v; rs += fabsf(v); }
      nrm = fmaxf(nrm, rs);
    }
    int sq = 0;
    while (nrm > 0.5f && sq < 24) { nrm *= 0.5f; ++sq; }
    const float sc = ldexpf(1.0f, -sq);
    for (int i = 0; i < 4; ++i) for (int j = 0; j < 4; ++j) T[i][j] *= sc;
    float E[4][4], P[4][4];
    for (int i = 0; i < 4; ++i)
      for (int j = 0; j < 4; ++j) {
        P[i][j] = T[i][j];
        E[i][j] = ((i == j) ? 1.0f : 0.0f) + T[i][j];
      }
    for (int k = 2; k <= 12; ++k) {
      float Pn[4][4];
      const float inv = 1.0f / (float)k;
      for (int i = 0; i < 4; ++i)
        for (int j = 0; j < 4; ++j) {
          float s = 0.0f;
          for (int m = 0; m < 4; ++m) s += P[i][m] * T[m][j];
          Pn[i][j] = s * inv;
        }
      for (int i = 0; i < 4; ++i)
        for (int j = 0; j < 4; ++j) { P[i][j] = Pn[i][j]; E[i][j] += Pn[i][j]; }
    }
    for (int s2 = 0; s2 < sq; ++s2) {
      float E2[4][4];
      for (int i = 0; i < 4; ++i)
        for (int j = 0; j < 4; ++j) {
          float s = 0.0f;
          for (int m = 0; m < 4; ++m) s += E[i][m] * E[m][j];
          E2[i][j] = s;
        }
      for (int i = 0; i < 4; ++i) for (int j = 0; j < 4; ++j) E[i][j] = E2[i][j];
    }
    // Fold Y-branch: G = Phi + w_out * r_in^T  =>  out = G @ xs
    for (int i = 0; i < 4; ++i)
      for (int j = 0; j < 4; ++j)
        Gm[tid * 16 + i * 4 + j] = E[i][j] + wout[i] * rin[j];
  }
  __syncthreads();

  // ---- Phase 4: out[row] = G @ xs  (x re-read hits WGP$/L2; coalesced f4 I/O)
  if (grow < Brows) {
    float g[16];
    for (int k = 0; k < 16; ++k) g[k] = Gm[row * 16 + k];
    const float* xrow = x + (size_t)grow * IN_DIM;
    float*       orow = out + (size_t)grow * IN_DIM;
    __builtin_prefetch(xrow + sub * 64, 0, 3);
    for (int k = 0; k < 16; ++k) {
      const int d4 = sub + 16 * k;     // float4 index within each 1024-block
      float4 xj[4];
      for (int j = 0; j < 4; ++j)
        xj[j] = *((const float4*)(xrow + j * BLOCK) + d4);
      for (int i = 0; i < 4; ++i) {
        float4 o;
        o.x = g[i*4+0]*xj[0].x + g[i*4+1]*xj[1].x + g[i*4+2]*xj[2].x + g[i*4+3]*xj[3].x;
        o.y = g[i*4+0]*xj[0].y + g[i*4+1]*xj[1].y + g[i*4+2]*xj[2].y + g[i*4+3]*xj[3].y;
        o.z = g[i*4+0]*xj[0].z + g[i*4+1]*xj[1].z + g[i*4+2]*xj[2].z + g[i*4+3]*xj[3].z;
        o.w = g[i*4+0]*xj[0].w + g[i*4+1]*xj[1].w + g[i*4+2]*xj[2].w + g[i*4+3]*xj[3].w;
        *((float4*)(orow + i * BLOCK) + d4) = o;
      }
    }
  }
}

extern "C" void kernel_launch(void* const* d_in, const int* in_sizes, int n_in,
                              void* d_out, int out_size, void* d_ws, size_t ws_size,
                              hipStream_t stream) {
  const float* x         = (const float*)d_in[0];
  const float* read_in   = (const float*)d_in[1];
  const float* alpha_ri  = (const float*)d_in[2];
  const float* write_out = (const float*)d_in[3];
  const float* alpha_wo  = (const float*)d_in[4];
  const float* W_ri      = (const float*)d_in[5];
  const float* W_wo      = (const float*)d_in[6];
  const float* log_dt    = (const float*)d_in[7];
  const float* W_dt      = (const float*)d_in[8];
  const float* b_dt      = (const float*)d_in[9];
  const float* conserv_A = (const float*)d_in[10];
  const float* W_conv    = (const float*)d_in[11];
  const float* b_conv    = (const float*)d_in[12];
  const float* diss_A    = (const float*)d_in[13];
  const float* W_diss    = (const float*)d_in[14];
  const float* b_diss    = (const float*)d_in[15];

  const int Brows = in_sizes[0] / IN_DIM;     // 16384
  bf16_t* wswz = (bf16_t*)d_ws;               // 128*3*32*16 bf16 = 384 KB

  const int swz_total = KTILES * NTILES * 32 * 16;
  cghc_weight_swizzle<<<(swz_total + 255) / 256, 256, 0, stream>>>(
      W_ri, W_wo, W_dt, W_conv, W_diss, wswz);

  const int tiles = (Brows + TILE_ROWS - 1) / TILE_ROWS;  // 1024
  cghc_kernel<<<tiles, 256, 0, stream>>>(
      x, read_in, alpha_ri, write_out, alpha_wo, log_dt, b_dt,
      conserv_A, b_conv, diss_A, b_diss, wswz, (float*)d_out, Brows);
}
